// MultiHeadAttention_54408645706047
// MI455X (gfx1250) — matrix operand order repaired and checked by reading, not compiled, not run on hardware
//
#include <hip/hip_runtime.h>
#include <hip/hip_bf16.h>

// Problem dims (fixed by the reference)
#define BB   8
#define SS   1024
#define DDIM 768
#define HH   12
#define DKV  64

typedef __bf16 bf16_t;
typedef __attribute__((address_space(3))) __bf16 lds_bf16;
typedef __attribute__((ext_vector_type(16))) __bf16 v16bf;
typedef __attribute__((ext_vector_type(8)))  __bf16 v8bf;
typedef __attribute__((ext_vector_type(8)))  float  v8f;

// float -> bf16, round-to-nearest-even
__device__ __forceinline__ bf16_t f2bf(float f) {
  unsigned u = __builtin_bit_cast(unsigned, f);
  unsigned r = (u + 0x7FFFu + ((u >> 16) & 1u)) >> 16;
  unsigned short s = (unsigned short)r;
  return __builtin_bit_cast(bf16_t, s);
}

// Async 16-byte global -> LDS transfer (CDNA5 GLOBAL_LOAD_ASYNC_TO_LDS_B128,
// tracked by ASYNCcnt; LDS written directly, no VGPR round trip).
__device__ __forceinline__ void async_ld16(bf16_t* lds_dst, const bf16_t* gsrc) {
  unsigned loff = (unsigned)(size_t)(lds_bf16*)lds_dst;  // LDS byte offset
  asm volatile("global_load_async_to_lds_b128 %0, %1, off"
               :: "v"(loff), "v"(gsrc) : "memory");
}
__device__ __forceinline__ void wait_async0() {
  asm volatile("s_wait_asynccnt 0" ::: "memory");
}

// A-fragment (16x32 bf16, row-major src, row stride ld elems).
// Contiguous k order: lanes 0-15 hold M=lane, K = k0..k0+15;
//             lanes 16-31 hold M=lane-16, K = k0+16..31.
__device__ __forceinline__ v16bf load_a_frag(const bf16_t* __restrict__ src,
                                             int ld, int row0, int k0, int lane) {
  int m = lane & 15, h = lane >> 4;
  const bf16_t* p = src + (size_t)(row0 + m) * ld + k0 + h * 16;
  v8bf lo = *(const v8bf*)p;          // K = k0 + h*16 .. +7
  v8bf hi = *(const v8bf*)(p + 8);    // K = k0 + h*16 + 8 .. +7
  v16bf r;
#pragma unroll
  for (int i = 0; i < 8; ++i) { r[i] = lo[i]; r[i + 8] = hi[i]; }
  return r;
}

// B-fragment (32x16 bf16) from TRANSPOSED storage srcT[n][k] (row stride ld).
// Contiguous k order: lanes 0-15 hold N=lane, K = k0..k0+15; lanes 16-31: K = k0+16..31.
__device__ __forceinline__ v16bf load_b_frag(const bf16_t* __restrict__ srcT,
                                             int ld, int n0, int k0, int lane) {
  const bf16_t* p = srcT + (size_t)(n0 + (lane & 15)) * ld + k0 + (lane >> 4) * 16;
  v8bf lo = *(const v8bf*)p;
  v8bf hi = *(const v8bf*)(p + 8);
  v16bf r;
#pragma unroll
  for (int i = 0; i < 8; ++i) { r[i] = lo[i]; r[i + 8] = hi[i]; }
  return r;
}

__device__ __forceinline__ v8f wmma_bf16(v16bf a, v16bf b, v8f c) {
  return __builtin_amdgcn_wmma_f32_16x16x32_bf16(false, a, false, b,
                                                 (short)0, c, false, false);
}

// ---------------- conversion kernels ----------------
__global__ void conv_flat_kernel(const float* __restrict__ in,
                                 bf16_t* __restrict__ out, long n) {
  for (long i = blockIdx.x * (long)blockDim.x + threadIdx.x; i < n;
       i += (long)gridDim.x * blockDim.x)
    out[i] = f2bf(in[i]);
}

// in[b][k][n] (K rows, N cols) -> out[b][n][k], with fp32->bf16 convert
__global__ void convT_kernel(const float* __restrict__ in, bf16_t* __restrict__ out,
                             int K, int N, long total) {
  long kn = (long)K * N;
  for (long i = blockIdx.x * (long)blockDim.x + threadIdx.x; i < total;
       i += (long)gridDim.x * blockDim.x) {
    long bm = i / kn;
    long r  = i - bm * kn;
    int  k  = (int)(r / N);
    int  n  = (int)(r - (long)k * N);
    out[bm * kn + (long)n * K + k] = f2bf(in[i]);
  }
}

// ---------------- QKV projection ----------------
// grid (S/128, B*H, 3), block 128 (4 waves).
// Each wave: 2 M-tiles (32 rows) x 64 cols -> every B-frag reused twice.
__global__ __launch_bounds__(128) void qkv_kernel(
    const bf16_t* __restrict__ xb,
    const bf16_t* __restrict__ WqT, const bf16_t* __restrict__ WkT,
    const bf16_t* __restrict__ WvT,
    const float* __restrict__ bq, const float* __restrict__ bk,
    const float* __restrict__ bv,
    bf16_t* __restrict__ Qo, bf16_t* __restrict__ Ko, bf16_t* __restrict__ VTo) {
  int which = blockIdx.z;
  int bh = blockIdx.y;
  int h  = bh % HH;
  int b  = bh / HH;
  int wave = threadIdx.x >> 5, lane = threadIdx.x & 31;
  int row0 = blockIdx.x * 128 + wave * 32;

  const bf16_t* A  = xb + (size_t)b * SS * DDIM;
  const bf16_t* Bt = (which == 0 ? WqT : which == 1 ? WkT : WvT) +
                     (size_t)h * DKV * DDIM;
  const float* bias = (which == 0 ? bq : which == 1 ? bk : bv) + h * DKV;

  v8f acc0[4] = {}, acc1[4] = {};
  for (int k0 = 0; k0 < DDIM; k0 += 32) {
    // group ALL loads of this k-step before the WMMA batch (partial waits)
    v16bf a0 = load_a_frag(A, DDIM, row0,      k0, lane);
    v16bf a1 = load_a_frag(A, DDIM, row0 + 16, k0, lane);
    v16bf bf[4];
#pragma unroll
    for (int c = 0; c < 4; ++c) bf[c] = load_b_frag(Bt, DDIM, c * 16, k0, lane);
#pragma unroll
    for (int c = 0; c < 4; ++c) {
      acc0[c] = wmma_bf16(a0, bf[c], acc0[c]);
      acc1[c] = wmma_bf16(a1, bf[c], acc1[c]);
    }
  }

  int half = lane >> 4, nlo = lane & 15;
#pragma unroll
  for (int mt = 0; mt < 2; ++mt) {
    v8f* acc = mt ? acc1 : acc0;
#pragma unroll
    for (int c = 0; c < 4; ++c) {
      int col = c * 16 + nlo;
      float bs = bias[col];
#pragma unroll
      for (int r = 0; r < 8; ++r) {
        int row = row0 + mt * 16 + half * 8 + r;
        float val = acc[c][r] + bs;
        if (which == 2)     // V stored transposed [dv][S] for contiguous PV B-frags
          VTo[((size_t)bh * DKV + col) * SS + row] = f2bf(val);
        else if (which == 0)
          Qo[((size_t)bh * SS + row) * DKV + col] = f2bf(val);
        else
          Ko[((size_t)bh * SS + row) * DKV + col] = f2bf(val);
      }
    }
  }
}

// ---------------- flash attention ----------------
// grid (S/64, B*H), block 128. Wave w owns q-rows [qt*64+w*16, +16).
// K/V tiles stream through double-buffered LDS via ASYNC global->LDS copies:
// tile j+1 is prefetched (ASYNCcnt) while tile j runs WMMA + softmax.
__global__ __launch_bounds__(128) void attn_kernel(
    const bf16_t* __restrict__ Q, const bf16_t* __restrict__ K,
    const bf16_t* __restrict__ VT, bf16_t* __restrict__ O) {
  __shared__ __align__(16) bf16_t kbuf[2][64 * 64];  // [key_local][dk]
  __shared__ __align__(16) bf16_t vbuf[2][64 * 64];  // [dv][key_local]
  __shared__ __align__(16) bf16_t pbuf[4][16 * 64];  // per-wave P tile

  int bh = blockIdx.y;
  int h  = bh % HH;
  int b  = bh / HH;
  int qt = blockIdx.x;
  int tid = threadIdx.x;
  int wave = tid >> 5, lane = tid & 31;
  int half = lane >> 4, nlo = lane & 15;
  int qrow0 = qt * 64 + wave * 16;

  const bf16_t* Qp = Q  + (size_t)bh * SS * DKV;
  const bf16_t* Kp = K  + (size_t)bh * SS * DKV;
  const bf16_t* Vp = VT + (size_t)bh * DKV * SS;
  bf16_t* pw = &pbuf[wave][0];

  // Q fragments live in registers for the whole row-tile
  v16bf qa0 = load_a_frag(Qp, DKV, qrow0, 0, lane);
  v16bf qa1 = load_a_frag(Qp, DKV, qrow0, 32, lane);

  float mrow[8], lrow[8];
  v8f o[4] = {};
#pragma unroll
  for (int r = 0; r < 8; ++r) { mrow[r] = -3.0e38f; lrow[r] = 0.f; }
  const float scale = 0.125f;  // 1/sqrt(64)

  // async-prefetch tile 0 (K tile is contiguous 8KB; V tile rows strided by S)
#pragma unroll
  for (int i = 0; i < 4; ++i) {
    int e = (tid + 128 * i) * 8;
    async_ld16(&kbuf[0][e], Kp + e);
  }
#pragma unroll
  for (int i = 0; i < 4; ++i) {
    int e = (tid + 128 * i) * 8;
    int row = e >> 6, col = e & 63;
    async_ld16(&vbuf[0][e], Vp + (size_t)row * SS + col);
  }

  for (int j = 0; j <= qt; ++j) {   // causal: only tiles touching lower triangle
    int kv0 = j * 64;
    int p = j & 1;

    wait_async0();     // tile j resident in LDS (per-wave; loads complete in order)
    __syncthreads();   // all waves ready; also fences last iter's buf[1-p] readers

    if (j < qt) {      // prefetch tile j+1 under tile j's compute
      const bf16_t* kg = Kp + (size_t)(kv0 + 64) * DKV;
#pragma unroll
      for (int i = 0; i < 4; ++i) {
        int e = (tid + 128 * i) * 8;
        async_ld16(&kbuf[1 - p][e], kg + e);
      }
#pragma unroll
      for (int i = 0; i < 4; ++i) {
        int e = (tid + 128 * i) * 8;
        int row = e >> 6, col = e & 63;
        async_ld16(&vbuf[1 - p][e], Vp + (size_t)row * SS + kv0 + 64 + col);
      }
    }

    // scores: S = Q * K^T  (B-frags from LDS, grouped before WMMA batch)
    v8f s[4] = {};
    {
      v16bf kb0[4], kb1[4];
#pragma unroll
      for (int c = 0; c < 4; ++c) {
        kb0[c] = load_b_frag(&kbuf[p][0], 64, c * 16, 0, lane);
        kb1[c] = load_b_frag(&kbuf[p][0], 64, c * 16, 32, lane);
      }
#pragma unroll
      for (int c = 0; c < 4; ++c) {
        s[c] = wmma_bf16(qa0, kb0[c], s[c]);
        s[c] = wmma_bf16(qa1, kb1[c], s[c]);
      }
    }

    bool diag = (j == qt);   // wave-uniform
#pragma unroll
    for (int c = 0; c < 4; ++c)
#pragma unroll
      for (int r = 0; r < 8; ++r) {
        float v = s[c][r] * scale;
        if (diag) {
          int qr = qrow0 + half * 8 + r;
          int kc = kv0 + c * 16 + nlo;
          v = (kc > qr) ? -3.0e38f : v;
        }
        s[c][r] = v;
      }

    // online softmax: per-row max/sum across 16 lanes of each half
#pragma unroll
    for (int r = 0; r < 8; ++r) {
      float mx = fmaxf(fmaxf(s[0][r], s[1][r]), fmaxf(s[2][r], s[3][r]));
#pragma unroll
      for (int off = 1; off < 16; off <<= 1)
        mx = fmaxf(mx, __shfl_xor(mx, off, 32));
      float nm = fmaxf(mrow[r], mx);
      float corr = __expf(mrow[r] - nm);
      mrow[r] = nm;
      float ps = 0.f;
#pragma unroll
      for (int c = 0; c < 4; ++c) {
        float pv = __expf(s[c][r] - nm);
        s[c][r] = pv;
        ps += pv;
      }
#pragma unroll
      for (int off = 1; off < 16; off <<= 1)
        ps += __shfl_xor(ps, off, 32);
      lrow[r] = lrow[r] * corr + ps;
#pragma unroll
      for (int c = 0; c < 4; ++c) o[c][r] *= corr;
    }

    // C-layout -> A-layout transpose of P through per-wave LDS tile
#pragma unroll
    for (int c = 0; c < 4; ++c)
#pragma unroll
      for (int r = 0; r < 8; ++r)
        pw[(half * 8 + r) * 64 + c * 16 + nlo] = f2bf(s[c][r]);
    asm volatile("s_wait_dscnt 0" ::: "memory");

    // O += P * V  (P A-frags + V B-frags all from LDS)
    {
      v16bf pa0 = load_a_frag(pw, 64, 0, 0, lane);
      v16bf pa1 = load_a_frag(pw, 64, 0, 32, lane);
      v16bf vb0[4], vb1[4];
#pragma unroll
      for (int c = 0; c < 4; ++c) {
        vb0[c] = load_b_frag(&vbuf[p][0], 64, c * 16, 0, lane);
        vb1[c] = load_b_frag(&vbuf[p][0], 64, c * 16, 32, lane);
      }
#pragma unroll
      for (int c = 0; c < 4; ++c) {
        o[c] = wmma_bf16(pa0, vb0[c], o[c]);
        o[c] = wmma_bf16(pa1, vb1[c], o[c]);
      }
    }
  }

  // normalize and store concat-head output [B,S,H*DV] (bf16)
#pragma unroll
  for (int c = 0; c < 4; ++c)
#pragma unroll
    for (int r = 0; r < 8; ++r) {
      int row = qrow0 + half * 8 + r;
      int dv  = c * 16 + nlo;
      float val = o[c][r] / lrow[r];
      O[((size_t)b * SS + row) * (HH * DKV) + h * DKV + dv] = f2bf(val);
    }
}

// ---------------- output projection ----------------
// grid (768/64, B*S/128), block 128. Each wave: 32 rows x 64 cols. fp32 out.
__global__ __launch_bounds__(128) void oproj_kernel(
    const bf16_t* __restrict__ A, const bf16_t* __restrict__ W0T,
    const float* __restrict__ b0, float* __restrict__ out) {
  int wave = threadIdx.x >> 5, lane = threadIdx.x & 31;
  int row0 = blockIdx.y * 128 + wave * 32;
  int n0   = blockIdx.x * 64;

  v8f acc0[4] = {}, acc1[4] = {};
  for (int k0 = 0; k0 < DDIM; k0 += 32) {
    v16bf a0 = load_a_frag(A, DDIM, row0,      k0, lane);
    v16bf a1 = load_a_frag(A, DDIM, row0 + 16, k0, lane);
    v16bf bf[4];
#pragma unroll
    for (int c = 0; c < 4; ++c)
      bf[c] = load_b_frag(W0T, DDIM, n0 + c * 16, k0, lane);
#pragma unroll
    for (int c = 0; c < 4; ++c) {
      acc0[c] = wmma_bf16(a0, bf[c], acc0[c]);
      acc1[c] = wmma_bf16(a1, bf[c], acc1[c]);
    }
  }

  int half = lane >> 4, nlo = lane & 15;
#pragma unroll
  for (int mt = 0; mt < 2; ++mt) {
    v8f* acc = mt ? acc1 : acc0;
#pragma unroll
    for (int c = 0; c < 4; ++c) {
      int col = n0 + c * 16 + nlo;
      float bs = b0[col];
#pragma unroll
      for (int r = 0; r < 8; ++r)
        out[(size_t)(row0 + mt * 16 + half * 8 + r) * DDIM + col] = acc[c][r] + bs;
    }
  }
}

extern "C" void kernel_launch(void* const* d_in, const int* in_sizes, int n_in,
                              void* d_out, int out_size, void* d_ws, size_t ws_size,
                              hipStream_t stream) {
  const float* x  = (const float*)d_in[0];
  const float* Wq = (const float*)d_in[1];
  const float* bq = (const float*)d_in[2];
  const float* Wk = (const float*)d_in[3];
  const float* bk = (const float*)d_in[4];
  const float* Wv = (const float*)d_in[5];
  const float* bv = (const float*)d_in[6];
  const float* W0 = (const float*)d_in[7];
  const float* b0 = (const float*)d_in[8];
  float* out = (float*)d_out;

  // workspace layout (bf16 elements), total ~64.5 MiB
  bf16_t* ws = (bf16_t*)d_ws;
  size_t off = 0;
  bf16_t* xb  = ws + off; off += (size_t)BB * SS * DDIM;      // x in bf16
  bf16_t* WqT = ws + off; off += (size_t)HH * DKV * DDIM;     // [H][64][768]
  bf16_t* WkT = ws + off; off += (size_t)HH * DKV * DDIM;
  bf16_t* WvT = ws + off; off += (size_t)HH * DKV * DDIM;
  bf16_t* W0T = ws + off; off += (size_t)DDIM * DDIM;         // [768][768]
  bf16_t* Qb  = ws + off; off += (size_t)BB * HH * SS * DKV;  // [bh][S][64]
  bf16_t* Kb  = ws + off; off += (size_t)BB * HH * SS * DKV;  // [bh][S][64]
  bf16_t* VTb = ws + off; off += (size_t)BB * HH * DKV * SS;  // [bh][64][S]
  bf16_t* AOb = ws + off; off += (size_t)BB * SS * HH * DKV;  // [B][S][768]

  conv_flat_kernel<<<1024, 256, 0, stream>>>(x, xb, (long)BB * SS * DDIM);
  convT_kernel<<<512, 256, 0, stream>>>(Wq, WqT, DDIM, DKV, (long)HH * DDIM * DKV);
  convT_kernel<<<512, 256, 0, stream>>>(Wk, WkT, DDIM, DKV, (long)HH * DDIM * DKV);
  convT_kernel<<<512, 256, 0, stream>>>(Wv, WvT, DDIM, DKV, (long)HH * DDIM * DKV);
  convT_kernel<<<512, 256, 0, stream>>>(W0, W0T, DDIM, DDIM, (long)DDIM * DDIM);

  dim3 g1(SS / 128, BB * HH, 3);
  qkv_kernel<<<g1, 128, 0, stream>>>(xb, WqT, WkT, WvT, bq, bk, bv, Qb, Kb, VTb);

  dim3 g2(SS / 64, BB * HH);
  attn_kernel<<<g2, 128, 0, stream>>>(Qb, Kb, VTb, AOb);

  dim3 g3(DDIM / 64, (BB * SS) / 128);
  oproj_kernel<<<g3, 128, 0, stream>>>(AOb, W0T, b0, out);
}
